// EmmaPolicyDecoder_25031069401318
// MI455X (gfx1250) — compile-verified, matrix operand order checked by reading, not run on hardware
//
#include <hip/hip_runtime.h>
#include <stdint.h>

typedef unsigned short u16;
typedef __attribute__((ext_vector_type(16))) __bf16 v16bf;
typedef __attribute__((ext_vector_type(8)))  float  v8f;

#define NB_B 32
#define NB_T 832
#define NB_D 256
#define NB_H 4
#define NB_F 1024
#define BT   (NB_B * NB_T)     // 26624
#define NQT  (NB_T / 16)       // 52 query tiles

// ---------- bf16 helpers ----------
__device__ __forceinline__ u16 f2bf(float x) {
  uint32_t u = __builtin_bit_cast(uint32_t, x);
  uint32_t r = u + 0x7FFFu + ((u >> 16) & 1u);
  return (u16)(r >> 16);
}
__device__ __forceinline__ float bf2f(u16 s) {
  uint32_t u = ((uint32_t)s) << 16;
  return __builtin_bit_cast(float, u);
}

// ---------- WMMA fragment load ----------
// A 16x32 bf16 fragment (per CDNA5 ISA layout): halves[0..7] = 8 contiguous
// bf16 at p, halves[8..15] = 8 contiguous bf16 at p+16 elements.
struct Frag { uint32_t u[8]; };

__device__ __forceinline__ v16bf ldfrag(const u16* p) {
  Frag f;
  *reinterpret_cast<uint4*>(&f.u[0]) = *reinterpret_cast<const uint4*>(p);
  *reinterpret_cast<uint4*>(&f.u[4]) = *reinterpret_cast<const uint4*>(p + 16);
  return __builtin_bit_cast(v16bf, f);
}

__device__ __forceinline__ v8f bfwmma(v16bf a, v16bf b, v8f c) {
  return __builtin_amdgcn_wmma_f32_16x16x32_bf16(false, a, false, b, (short)0, c,
                                                 false, false);
}

__device__ __forceinline__ float gelu_f(float x) {
  float x3 = x * x * x;
  return 0.5f * x * (1.f + tanhf(0.7978845608028654f * (x + 0.044715f * x3)));
}

// ---------- weight transpose fp32 [K,N] -> bf16 [N,K] ----------
__global__ __launch_bounds__(256) void transpose_bf16_kernel(
    const float* __restrict__ src, u16* __restrict__ dst, int K, int N) {
  int i = blockIdx.x * 256 + threadIdx.x;
  if (i >= K * N) return;
  int k = i / N, n = i - k * N;
  dst[(size_t)n * K + k] = f2bf(src[i]);
}

// ---------- generic WMMA GEMM (2-deep ping-pong pipeline, K % 64 == 0) ----
// C[M,N] = act(A_bf16[M,K] @ WT_bf16[N,K]^T + bias)
enum GemmMode { M_F32ACC = 0, M_GELU = 1, M_LEAKY = 2, M_QK = 3, M_VT = 4 };

__global__ __launch_bounds__(128) void gemm_wmma_kernel(
    const u16* __restrict__ A, const u16* __restrict__ WT,
    const float* __restrict__ bias, int M, int N, int K, int mode,
    float* __restrict__ Cf, u16* __restrict__ Cb, int Tdim) {
  const int lane = threadIdx.x & 31;
  const int wave = threadIdx.x >> 5;
  const int m0 = (blockIdx.x * 4 + wave) * 16;
  const int n0 = blockIdx.y * 64;
  if (m0 >= M) return;  // wave-uniform
  const int kb = (lane >> 4) * 8;
  const int lm = lane & 15;
  const u16* Ap = A + (size_t)(m0 + lm) * K + kb;
  const u16* Bp[4];
#pragma unroll
  for (int nt = 0; nt < 4; nt++)
    Bp[nt] = WT + (size_t)(n0 + nt * 16 + lm) * K + kb;

  v8f acc[4] = {};
  // prologue: two k-steps in flight (k = 0 and k = 32)
  v16bf a0 = ldfrag(Ap);
  v16bf b0[4];
#pragma unroll
  for (int nt = 0; nt < 4; nt++) b0[nt] = ldfrag(Bp[nt]);
  v16bf a1 = ldfrag(Ap + 32);
  v16bf b1[4];
#pragma unroll
  for (int nt = 0; nt < 4; nt++) b1[nt] = ldfrag(Bp[nt] + 32);

  int k0 = 0;
  for (; k0 < K - 64; k0 += 64) {
    // consume buffer 0 (k0), immediately refill it for k0+64
#pragma unroll
    for (int nt = 0; nt < 4; nt++) acc[nt] = bfwmma(a0, b0[nt], acc[nt]);
    __builtin_prefetch(Ap + k0 + 192, 0, 3);
    a0 = ldfrag(Ap + k0 + 64);
#pragma unroll
    for (int nt = 0; nt < 4; nt++) {
      __builtin_prefetch(Bp[nt] + k0 + 192, 0, 3);
      b0[nt] = ldfrag(Bp[nt] + k0 + 64);
    }
    // consume buffer 1 (k0+32), refill for k0+96
#pragma unroll
    for (int nt = 0; nt < 4; nt++) acc[nt] = bfwmma(a1, b1[nt], acc[nt]);
    a1 = ldfrag(Ap + k0 + 96);
#pragma unroll
    for (int nt = 0; nt < 4; nt++) b1[nt] = ldfrag(Bp[nt] + k0 + 96);
  }
  // drain the two in-flight buffers (k = K-64 and K-32)
#pragma unroll
  for (int nt = 0; nt < 4; nt++) acc[nt] = bfwmma(a0, b0[nt], acc[nt]);
#pragma unroll
  for (int nt = 0; nt < 4; nt++) acc[nt] = bfwmma(a1, b1[nt], acc[nt]);

#pragma unroll
  for (int nt = 0; nt < 4; nt++) {
    int n = n0 + nt * 16 + lm;
    float bi = bias ? bias[n] : 0.f;
#pragma unroll
    for (int j = 0; j < 8; j++) {
      int m = m0 + (lane >> 4) * 8 + j;
      float v = acc[nt][j] + bi;
      switch (mode) {
        case M_F32ACC:
          Cf[(size_t)m * N + n] += v;
          break;
        case M_GELU:
          Cb[(size_t)m * N + n] = f2bf(gelu_f(v));
          break;
        case M_LEAKY:
          Cb[(size_t)m * N + n] = f2bf(v > 0.f ? v : 0.01f * v);
          break;
        case M_QK: {
          int b = m / Tdim, t = m - b * Tdim;
          int h = n >> 6, hd = n & 63;
          Cb[(((size_t)b * NB_H + h) * Tdim + t) * 64 + hd] = f2bf(v);
          break;
        }
        case M_VT: {
          int b = m / Tdim, t = m - b * Tdim;
          int h = n >> 6, hd = n & 63;
          Cb[(((size_t)b * NB_H + h) * 64 + hd) * Tdim + t] = f2bf(v);
          break;
        }
      }
    }
  }
}

// ---------- flash attention (1 wave per (b,h,qtile)) ----------
__global__ __launch_bounds__(32) void attn_kernel(
    const u16* __restrict__ Qb, const u16* __restrict__ Kb,
    const u16* __restrict__ Vt, u16* __restrict__ Xo) {
  const int lane = threadIdx.x;
  int idx = blockIdx.x;
  int qt = idx % NQT;
  idx /= NQT;
  int h = idx & 3, b = idx >> 2;
  int bh = (b << 2) | h;
  const u16* Qp = Qb + ((size_t)bh * NB_T + qt * 16) * 64;
  const u16* Kp = Kb + (size_t)bh * NB_T * 64;
  const u16* Vp = Vt + (size_t)bh * 64 * NB_T;

  __shared__ float Sld[16][32];
  __shared__ u16 Pld[16][32];
  __shared__ float rowm[16], rowl[16], rowc[16];

  const int kb = (lane >> 4) * 8;
  const int lm = lane & 15;
  if (lane < 16) {
    rowm[lane] = -1e30f;
    rowl[lane] = 0.f;
  }
  __syncthreads();

  v16bf aq0 = ldfrag(Qp + (size_t)lm * 64 + kb);
  v16bf aq1 = ldfrag(Qp + (size_t)lm * 64 + 32 + kb);
  v8f o[4] = {};

  int nchunk = qt / 2 + 1;  // 32 keys per chunk
  for (int c = 0; c < nchunk; c++) {
    // Issue ALL loads for this chunk up front: 4 K-fragments (2 score tiles)
    // and 4 V-fragments (PV stage). The V loads are independent of the score
    // computation and the scalar softmax, so they overlap both.
    v16bf bk[4];
#pragma unroll
    for (int t2 = 0; t2 < 2; t2++) {
      int kt = 2 * c + t2;
      bk[2 * t2 + 0] = ldfrag(Kp + (size_t)(kt * 16 + lm) * 64 + kb);
      bk[2 * t2 + 1] = ldfrag(Kp + (size_t)(kt * 16 + lm) * 64 + 32 + kb);
    }
    v16bf bv[4];
#pragma unroll
    for (int nt = 0; nt < 4; nt++)
      bv[nt] = ldfrag(Vp + (size_t)(nt * 16 + lm) * NB_T + c * 32 + kb);

#pragma unroll
    for (int t2 = 0; t2 < 2; t2++) {
      int kt = 2 * c + t2;
      v8f s = {};
      s = bfwmma(aq0, bk[2 * t2 + 0], s);
      s = bfwmma(aq1, bk[2 * t2 + 1], s);
#pragma unroll
      for (int j = 0; j < 8; j++) {
        int m = (lane >> 4) * 8 + j;
        float v = s[j] * 0.125f;  // 1/sqrt(64)
        if (kt * 16 + lm > qt * 16 + m) v = -1e9f;  // causal mask
        Sld[m][t2 * 16 + lm] = v;
      }
    }
    __syncthreads();
    if (lane < 16) {
      int r = lane;
      float mold = rowm[r], mnew = mold;
      for (int n = 0; n < 32; n++) mnew = fmaxf(mnew, Sld[r][n]);
      float corr = __expf(mold - mnew);
      float sum = 0.f;
      for (int n = 0; n < 32; n++) {
        float p = __expf(Sld[r][n] - mnew);
        Pld[r][n] = f2bf(p);
        sum += p;
      }
      rowl[r] = rowl[r] * corr + sum;
      rowm[r] = mnew;
      rowc[r] = corr;
    }
    __syncthreads();
#pragma unroll
    for (int nt = 0; nt < 4; nt++)
#pragma unroll
      for (int j = 0; j < 8; j++) o[nt][j] *= rowc[(lane >> 4) * 8 + j];

    v16bf ap = ldfrag(&Pld[lm][kb]);
#pragma unroll
    for (int nt = 0; nt < 4; nt++) o[nt] = bfwmma(ap, bv[nt], o[nt]);
    __syncthreads();
  }
#pragma unroll
  for (int nt = 0; nt < 4; nt++)
#pragma unroll
    for (int j = 0; j < 8; j++) {
      int m = (lane >> 4) * 8 + j;
      float v = o[nt][j] / rowl[m];
      int t = qt * 16 + m;
      Xo[((size_t)b * NB_T + t) * NB_D + h * 64 + nt * 16 + lm] = f2bf(v);
    }
}

// ---------- LayerNorm (D=256) -> bf16 ----------
__global__ __launch_bounds__(256) void ln_kernel(
    const float* __restrict__ X, const float* __restrict__ sc,
    const float* __restrict__ bi, u16* __restrict__ out) {
  int row = blockIdx.x, d = threadIdx.x;
  __shared__ float red[256];
  float v = X[(size_t)row * 256 + d];
  red[d] = v;
  __syncthreads();
  for (int o = 128; o > 0; o >>= 1) {
    if (d < o) red[d] += red[d + o];
    __syncthreads();
  }
  float mean = red[0] * (1.f / 256.f);
  __syncthreads();
  float c = v - mean;
  red[d] = c * c;
  __syncthreads();
  for (int o = 128; o > 0; o >>= 1) {
    if (d < o) red[d] += red[d + o];
    __syncthreads();
  }
  float var = red[0] * (1.f / 256.f);
  float y = c * rsqrtf(var + 1e-5f) * sc[d] + bi[d];
  out[(size_t)row * 256 + d] = f2bf(y);
}

// ---------- embeddings ----------
__global__ __launch_bounds__(256) void embed_kernel(
    const int* __restrict__ tokens, const float* __restrict__ ea,
    const float* __restrict__ eo, float* __restrict__ x) {
  int i = blockIdx.x * 256 + threadIdx.x;  // < B*T*D
  int d = i & 255, bt = i >> 8;
  int t = bt % NB_T;
  int tok = tokens[bt];
  const float* src = (t % 13 == 0) ? ea : eo;
  x[i] = src[tok * 256 + d];
}

__global__ __launch_bounds__(256) void addpos_kernel(
    float* __restrict__ x, const float* __restrict__ pe) {
  int i = blockIdx.x * 256 + threadIdx.x;
  int d = i & 255, bt = i >> 8;
  x[i] += pe[(bt % NB_T) * 256 + d];
}

// ---------- manual k/v softmax pooling over LM=36 ----------
__global__ __launch_bounds__(256) void manual_kv_kernel(
    const float* __restrict__ em, const float* __restrict__ mkw,
    const float* __restrict__ mkb, const float* __restrict__ mvw,
    const float* __restrict__ mvb, float* __restrict__ mkeys,
    float* __restrict__ mvals) {
  int bs = blockIdx.x, d = threadIdx.x;
  const float* E = em + (size_t)bs * 36 * 256;
  __shared__ float redk[256], redv[256];
  __shared__ float sk[36], sv[36];
  float wk = mkw[d], wv = mvw[d];
  for (int l = 0; l < 36; l++) {
    float e = E[l * 256 + d];
    redk[d] = e * wk;
    redv[d] = e * wv;
    __syncthreads();
    for (int o = 128; o > 0; o >>= 1) {
      if (d < o) {
        redk[d] += redk[d + o];
        redv[d] += redv[d + o];
      }
      __syncthreads();
    }
    if (d == 0) {
      sk[l] = redk[0] + mkb[0];
      sv[l] = redv[0] + mvb[0];
    }
    __syncthreads();
  }
  if (d == 0) {
    float mk = -1e30f, mv = -1e30f;
    for (int l = 0; l < 36; l++) {
      mk = fmaxf(mk, sk[l]);
      mv = fmaxf(mv, sv[l]);
    }
    float s1 = 0.f, s2 = 0.f;
    for (int l = 0; l < 36; l++) {
      sk[l] = __expf(sk[l] - mk);
      sv[l] = __expf(sv[l] - mv);
      s1 += sk[l];
      s2 += sv[l];
    }
    for (int l = 0; l < 36; l++) {
      sk[l] /= s1;
      sv[l] /= s2;
    }
  }
  __syncthreads();
  float aK = 0.f, aV = 0.f;
  for (int l = 0; l < 36; l++) {
    float e = E[l * 256 + d];
    aK += sk[l] * e;
    aV += sv[l] * e;
  }
  mkeys[bs * 256 + d] = aK;
  mvals[bs * 256 + d] = aV;
}

// ---------- project manual keys/values ----------
__global__ __launch_bounds__(256) void manual_proj_kernel(
    const float* __restrict__ mkeys, const float* __restrict__ mvals,
    const float* __restrict__ wk, const float* __restrict__ bk,
    const float* __restrict__ wv, const float* __restrict__ bv,
    float* __restrict__ kproj, float* __restrict__ vproj) {
  int bs = blockIdx.x, d = threadIdx.x;
  __shared__ float rk[256], rv[256];
  rk[d] = mkeys[bs * 256 + d];
  rv[d] = mvals[bs * 256 + d];
  __syncthreads();
  float ak = bk[d], av = bv[d];
  for (int k = 0; k < 256; k++) {
    ak += rk[k] * wk[k * 256 + d];
    av += rv[k] * wv[k * 256 + d];
  }
  kproj[bs * 256 + d] = ak;
  vproj[bs * 256 + d] = av;
}

// ---------- manual cross-attention (3 keys), adds into x at Q positions ----------
__global__ __launch_bounds__(256) void manual_attn_kernel(
    float* __restrict__ x, const float* __restrict__ wq,
    const float* __restrict__ bq, const float* __restrict__ kproj,
    const float* __restrict__ vproj, const float* __restrict__ wo,
    const float* __restrict__ bo) {
  int idx = blockIdx.x;  // b*192 + qi
  int b = idx / 192, qi = idx - b * 192;
  int qpos = (qi / 3) * 13 + 1 + (qi % 3) * 3;  // offsets 1,4,7
  int d = threadIdx.x;
  __shared__ float er[256], qr[256], red[256], avr[256], p[3];
  er[d] = x[((size_t)b * NB_T + qpos) * 256 + d];
  __syncthreads();
  float q = bq[d];
  for (int k = 0; k < 256; k++) q += er[k] * wq[k * 256 + d];
  qr[d] = q;
  __syncthreads();
  for (int si = 0; si < 3; si++) {
    red[d] = qr[d] * kproj[(b * 3 + si) * 256 + d];
    __syncthreads();
    for (int o = 128; o > 0; o >>= 1) {
      if (d < o) red[d] += red[d + o];
      __syncthreads();
    }
    if (d == 0) p[si] = red[0] * 0.0625f;  // 1/sqrt(256)
    __syncthreads();
  }
  if (d == 0) {
    float mx = fmaxf(p[0], fmaxf(p[1], p[2]));
    float s = 0.f;
    for (int si = 0; si < 3; si++) {
      p[si] = __expf(p[si] - mx);
      s += p[si];
    }
    for (int si = 0; si < 3; si++) p[si] /= s;
  }
  __syncthreads();
  avr[d] = p[0] * vproj[(b * 3 + 0) * 256 + d] +
           p[1] * vproj[(b * 3 + 1) * 256 + d] +
           p[2] * vproj[(b * 3 + 2) * 256 + d];
  __syncthreads();
  float delta = bo[d];
  for (int k = 0; k < 256; k++) delta += avr[k] * wo[k * 256 + d];
  x[((size_t)b * NB_T + qpos) * 256 + d] += delta;
}

// ---------- gather rows at positions t%13==12 ----------
__global__ __launch_bounds__(256) void gather_kernel(
    const u16* __restrict__ hbuf, u16* __restrict__ hx) {
  int row = blockIdx.x, d = threadIdx.x;  // row in [0,2048)
  int b = row >> 6, i = row & 63;
  int t = 13 * i + 12;
  hx[(size_t)row * 256 + d] = hbuf[((size_t)b * NB_T + t) * 256 + d];
}

// ---------- final heads: 5-way softmax + value ----------
__global__ __launch_bounds__(256) void head_final_kernel(
    const u16* __restrict__ ha2, const u16* __restrict__ hv2,
    const float* __restrict__ haw3, const float* __restrict__ hab3,
    const float* __restrict__ hvw3, const float* __restrict__ hvb3,
    float* __restrict__ out) {
  int row = blockIdx.x, d = threadIdx.x;  // row in [0,2048)
  __shared__ float red[256][6];
  float a = bf2f(ha2[(size_t)row * 256 + d]);
  float v = bf2f(hv2[(size_t)row * 256 + d]);
#pragma unroll
  for (int j = 0; j < 5; j++) red[d][j] = a * haw3[d * 5 + j];
  red[d][5] = v * hvw3[d];
  __syncthreads();
  for (int o = 128; o > 0; o >>= 1) {
    if (d < o)
#pragma unroll
      for (int j = 0; j < 6; j++) red[d][j] += red[d + o][j];
    __syncthreads();
  }
  if (d == 0) {
    float lg[5], mx = -1e30f, s = 0.f;
#pragma unroll
    for (int j = 0; j < 5; j++) {
      lg[j] = red[0][j] + hab3[j];
      mx = fmaxf(mx, lg[j]);
    }
#pragma unroll
    for (int j = 0; j < 5; j++) {
      lg[j] = __expf(lg[j] - mx);
      s += lg[j];
    }
#pragma unroll
    for (int j = 0; j < 5; j++) out[(size_t)row * 5 + j] = lg[j] / s;
    out[2048 * 5 + row] = red[0][5] + hvb3[0];
  }
}

// ================= host orchestration =================
extern "C" void kernel_launch(void* const* d_in, const int* in_sizes, int n_in,
                              void* d_out, int out_size, void* d_ws,
                              size_t ws_size, hipStream_t stream) {
  (void)in_sizes; (void)n_in; (void)out_size; (void)ws_size;
  const int* tokens = (const int*)d_in[0];
  const float* eman = (const float*)d_in[1];
  const float* emb_a = (const float*)d_in[2];
  const float* emb_o = (const float*)d_in[3];
  const float* pos = (const float*)d_in[4];
  const float* mk_w = (const float*)d_in[5];
  const float* mk_b = (const float*)d_in[6];
  const float* mv_w = (const float*)d_in[7];
  const float* mv_b = (const float*)d_in[8];
  const float* ma_wq = (const float*)d_in[9];
  const float* ma_bq = (const float*)d_in[10];
  const float* ma_wk = (const float*)d_in[11];
  const float* ma_bk = (const float*)d_in[12];
  const float* ma_wv = (const float*)d_in[13];
  const float* ma_bv = (const float*)d_in[14];
  const float* ma_wo = (const float*)d_in[15];
  const float* ma_bo = (const float*)d_in[16];
  const float* ln1_s = (const float*)d_in[17];
  const float* ln1_b = (const float*)d_in[18];
  const float* wq = (const float*)d_in[19];
  const float* bq = (const float*)d_in[20];
  const float* wk = (const float*)d_in[21];
  const float* bk = (const float*)d_in[22];
  const float* wv = (const float*)d_in[23];
  const float* bv = (const float*)d_in[24];
  const float* wo = (const float*)d_in[25];
  const float* bo = (const float*)d_in[26];
  const float* ln2_s = (const float*)d_in[27];
  const float* ln2_b = (const float*)d_in[28];
  const float* w1 = (const float*)d_in[29];
  const float* b1 = (const float*)d_in[30];
  const float* w2 = (const float*)d_in[31];
  const float* b2 = (const float*)d_in[32];
  const float* lnf_s = (const float*)d_in[33];
  const float* lnf_b = (const float*)d_in[34];
  const float* ha_w1 = (const float*)d_in[35];
  const float* ha_b1 = (const float*)d_in[36];
  const float* ha_w2 = (const float*)d_in[37];
  const float* ha_b2 = (const float*)d_in[38];
  const float* ha_w3 = (const float*)d_in[39];
  const float* ha_b3 = (const float*)d_in[40];
  const float* hv_w1 = (const float*)d_in[41];
  const float* hv_b1 = (const float*)d_in[42];
  const float* hv_w2 = (const float*)d_in[43];
  const float* hv_b2 = (const float*)d_in[44];
  const float* hv_w3 = (const float*)d_in[45];
  const float* hv_b3 = (const float*)d_in[46];

  // workspace bump allocator
  char* base = (char*)d_ws;
  size_t off = 0;
  auto alloc = [&](size_t elems, size_t esz) -> void* {
    void* p = base + off;
    off += (elems * esz + 255) & ~(size_t)255;
    return p;
  };
  float* x = (float*)alloc((size_t)BT * 256, 4);
  float* mkeys = (float*)alloc(96 * 256, 4);
  float* mvals = (float*)alloc(96 * 256, 4);
  float* kproj = (float*)alloc(96 * 256, 4);
  float* vproj = (float*)alloc(96 * 256, 4);
  u16* hbuf = (u16*)alloc((size_t)BT * 256, 2);
  u16* qbuf = (u16*)alloc((size_t)BT * 256, 2);
  u16* kbuf = (u16*)alloc((size_t)BT * 256, 2);
  u16* vbuf = (u16*)alloc((size_t)BT * 256, 2);
  u16* xatt = (u16*)alloc((size_t)BT * 256, 2);
  u16* t1 = (u16*)alloc((size_t)BT * 1024, 2);
  u16* wT = (u16*)alloc(262144, 2);
  u16* hx = (u16*)alloc(2048 * 256, 2);
  u16* h1 = (u16*)alloc(2048 * 256, 2);
  u16* a2 = (u16*)alloc(2048 * 256, 2);
  u16* v2 = (u16*)alloc(2048 * 256, 2);

  // ---- embeddings + manual attention ----
  manual_kv_kernel<<<96, 256, 0, stream>>>(eman, mk_w, mk_b, mv_w, mv_b, mkeys,
                                           mvals);
  manual_proj_kernel<<<96, 256, 0, stream>>>(mkeys, mvals, ma_wk, ma_bk, ma_wv,
                                             ma_bv, kproj, vproj);
  embed_kernel<<<BT, 256, 0, stream>>>(tokens, emb_a, emb_o, x);
  manual_attn_kernel<<<NB_B * 192, 256, 0, stream>>>(x, ma_wq, ma_bq, kproj,
                                                     vproj, ma_wo, ma_bo);
  addpos_kernel<<<BT, 256, 0, stream>>>(x, pos);

  dim3 g256(BT / 64, 256 / 64);    // M=26624, N=256
  dim3 g1024(BT / 64, 1024 / 64);  // M=26624, N=1024
  // ---- transformer layers ----
  for (int l = 0; l < 4; l++) {
    ln_kernel<<<BT, 256, 0, stream>>>(x, ln1_s + l * 256, ln1_b + l * 256, hbuf);

    transpose_bf16_kernel<<<256, 256, 0, stream>>>(wq + (size_t)l * 65536, wT, 256, 256);
    gemm_wmma_kernel<<<g256, 128, 0, stream>>>(hbuf, wT, bq + l * 256, BT, 256,
                                               256, M_QK, nullptr, qbuf, NB_T);
    transpose_bf16_kernel<<<256, 256, 0, stream>>>(wk + (size_t)l * 65536, wT, 256, 256);
    gemm_wmma_kernel<<<g256, 128, 0, stream>>>(hbuf, wT, bk + l * 256, BT, 256,
                                               256, M_QK, nullptr, kbuf, NB_T);
    transpose_bf16_kernel<<<256, 256, 0, stream>>>(wv + (size_t)l * 65536, wT, 256, 256);
    gemm_wmma_kernel<<<g256, 128, 0, stream>>>(hbuf, wT, bv + l * 256, BT, 256,
                                               256, M_VT, nullptr, vbuf, NB_T);

    attn_kernel<<<NB_B * NB_H * NQT, 32, 0, stream>>>(qbuf, kbuf, vbuf, xatt);

    transpose_bf16_kernel<<<256, 256, 0, stream>>>(wo + (size_t)l * 65536, wT, 256, 256);
    gemm_wmma_kernel<<<g256, 128, 0, stream>>>(xatt, wT, bo + l * 256, BT, 256,
                                               256, M_F32ACC, x, nullptr, NB_T);

    ln_kernel<<<BT, 256, 0, stream>>>(x, ln2_s + l * 256, ln2_b + l * 256, hbuf);
    transpose_bf16_kernel<<<1024, 256, 0, stream>>>(w1 + (size_t)l * 262144, wT, 256, 1024);
    gemm_wmma_kernel<<<g1024, 128, 0, stream>>>(hbuf, wT, b1 + l * 1024, BT,
                                                1024, 256, M_GELU, nullptr, t1, NB_T);
    transpose_bf16_kernel<<<1024, 256, 0, stream>>>(w2 + (size_t)l * 262144, wT, 1024, 256);
    gemm_wmma_kernel<<<g256, 128, 0, stream>>>(t1, wT, b2 + l * 256, BT, 256,
                                               1024, M_F32ACC, x, nullptr, NB_T);
  }

  // ---- final LN + heads ----
  ln_kernel<<<BT, 256, 0, stream>>>(x, lnf_s, lnf_b, hbuf);
  gather_kernel<<<2048, 256, 0, stream>>>(hbuf, hx);

  dim3 gh(2048 / 64, 256 / 64);
  transpose_bf16_kernel<<<256, 256, 0, stream>>>(ha_w1, wT, 256, 256);
  gemm_wmma_kernel<<<gh, 128, 0, stream>>>(hx, wT, ha_b1, 2048, 256, 256,
                                           M_LEAKY, nullptr, h1, NB_T);
  transpose_bf16_kernel<<<256, 256, 0, stream>>>(ha_w2, wT, 256, 256);
  gemm_wmma_kernel<<<gh, 128, 0, stream>>>(h1, wT, ha_b2, 2048, 256, 256,
                                           M_LEAKY, nullptr, a2, NB_T);
  transpose_bf16_kernel<<<256, 256, 0, stream>>>(hv_w1, wT, 256, 256);
  gemm_wmma_kernel<<<gh, 128, 0, stream>>>(hx, wT, hv_b1, 2048, 256, 256,
                                           M_LEAKY, nullptr, h1, NB_T);
  transpose_bf16_kernel<<<256, 256, 0, stream>>>(hv_w2, wT, 256, 256);
  gemm_wmma_kernel<<<gh, 128, 0, stream>>>(h1, wT, hv_b2, 2048, 256, 256,
                                           M_LEAKY, nullptr, v2, NB_T);

  head_final_kernel<<<2048, 256, 0, stream>>>(a2, v2, ha_w3, ha_b3, hv_w3,
                                              hv_b3, (float*)d_out);
}
